// IMPALA_78658031059153
// MI455X (gfx1250) — matrix operand order repaired and checked by reading, not compiled, not run on hardware
//
#include <hip/hip_runtime.h>
#include <hip/hip_bf16.h>

// ---------------------------------------------------------------------------
// IMPALA forward pass for MI455X (gfx1250, wave32, WMMA).
// All matmul-shaped work goes through v_wmma_f32_16x16x32_f16.
// Convs = implicit GEMM with im2col staged in LDS (one image per workgroup),
// staged via the Tensor Data Mover where the builtin is available.
// Weights pre-swizzled into exact WMMA B-fragment VGPR order.
// Dual/quad independent accumulator chains per wave hide the WMMA->WMMA
// RAW hazard (1 NOP + 4 co-exec slots on F16 WMMA).
// LSTM recurrence = single persistent workgroup with h/c/gates in 176KB LDS.
// ---------------------------------------------------------------------------

typedef __attribute__((ext_vector_type(16))) _Float16 v16h;
typedef __attribute__((ext_vector_type(8)))  float    v8f;

static __device__ __forceinline__ v8f wmma32f16(v16h a, v16h b, v8f c) {
    return __builtin_amdgcn_wmma_f32_16x16x32_f16(
        false, a, false, b, (short)0, c, false, false);
}

// B fragment, pre-swizzled in memory: frag f occupies 512 halves;
// lane L holds 16 contiguous halves at fragBase + L*16.
static __device__ __forceinline__ v16h load_b_frag(const _Float16* __restrict__ fragBase) {
    int lane = (int)(threadIdx.x & 31u);
    union { v16h v; uint4 q[2]; } r;
    const uint4* p = (const uint4*)(fragBase + lane * 16);
    r.q[0] = p[0];
    r.q[1] = p[1];
    return r.v;
}

// A fragment from an LDS row-major tile (16-bit A 16x32 layout: two contiguous
// 8-half runs per lane).
static __device__ __forceinline__ v16h load_a_frag_lds(const _Float16* sA, int lda,
                                                       int mBase, int kOff) {
    int lane  = (int)(threadIdx.x & 31u);
    int row   = mBase + (lane & 15);
    int khalf = (lane >> 4) * 8;
    union { v16h v; unsigned int u[8]; } r;
    const unsigned int* p0 = (const unsigned int*)(sA + row * lda + kOff + khalf);
    const unsigned int* p1 = (const unsigned int*)(sA + row * lda + kOff + 16 + khalf);
#pragma unroll
    for (int i = 0; i < 4; ++i) { r.u[i] = p0[i]; r.u[4 + i] = p1[i]; }
    return r.v;
}

static __device__ __forceinline__ float lrelu(float v) { return v > 0.f ? v : 0.01f * v; }
static __device__ __forceinline__ float sigm(float x)  { return 1.f / (1.f + expf(-x)); }

// ---------------------------------------------------------------------------
// Tensor Data Mover: linear global->LDS copy of `qwords` 8-byte units.
// D# per CDNA5 ISA ch.8: group0 = {count=1, lds_addr, global_addr, type=2},
// group1 = {data_size=8B, tensor_dim0=qwords, dim1=1, tile_dim0=qwords,
// tile_dim1=1, stride0=qwords}. Groups 2/3 zero (<=2D tensor).
// This toolchain's builtin arity (clang-23 / therock-10.0):
//   (u32x4 g0, i32x8 g1, i32x4 g2, i32x4 g3, i32x8, i32 cpol)
// ---------------------------------------------------------------------------
#if __has_builtin(__builtin_amdgcn_tensor_load_to_lds)
#define USE_TDM 1
typedef unsigned int u32x4_t __attribute__((ext_vector_type(4)));
typedef int          i32x8_t __attribute__((ext_vector_type(8)));
typedef int          i32x4_t __attribute__((ext_vector_type(4)));

static __device__ __forceinline__ void tdm_load_linear(void* ldsDst, const void* gsrc,
                                                       unsigned qwords /* <= 65535 */) {
    unsigned long long ga = (unsigned long long)(size_t)gsrc;
    unsigned lds = (unsigned)(size_t)ldsDst; // flat-shared addr[31:0] == LDS offset
    u32x4_t g0;
    g0[0] = 1u;                                   // count=1, user descriptor
    g0[1] = lds;                                  // lds_addr (bytes)
    g0[2] = (unsigned)ga;                         // global_addr[31:0]
    g0[3] = (unsigned)(ga >> 32) | (2u << 30);    // global_addr[56:32] | type=2
    i32x8_t g1;
    g1[0] = (int)(3u << 16);                      // data_size = 8 bytes
    g1[1] = (int)((qwords & 0xFFFFu) << 16);      // tensor_dim0[15:0]
    g1[2] = (int)(((qwords >> 16) & 0xFFFFu) | (1u << 16)); // dim0[31:16] | tensor_dim1=1
    g1[3] = (int)((qwords & 0xFFFFu) << 16);      // tile_dim0
    g1[4] = 1;                                    // tile_dim1 = 1 (tile_dim2 unused)
    g1[5] = (int)qwords;                          // tensor_dim0_stride[31:0]
    g1[6] = 0;
    g1[7] = 0;
    i32x4_t z4 = {0, 0, 0, 0};
    i32x8_t z8 = {0, 0, 0, 0, 0, 0, 0, 0};
    __builtin_amdgcn_tensor_load_to_lds(g0, g1, z4, z4, z8, 0);
}
#else
#define USE_TDM 0
#endif

// ---------------------------------------------------------------------------
// Weight prep: f32 [N][Ksrc-order] -> f16 B matrix [K][N] in our im2col K-order
// mode 0: src = n*Ksrc + k (k >= Ksrc -> 0)
// mode 1: k = hw*C + c -> src = n*Ksrc + c*HW + hw
// ---------------------------------------------------------------------------
__global__ void prep_b_kernel(const float* __restrict__ src, _Float16* __restrict__ Bm,
                              int K, int N, int Ksrc, int C, int HW, int mode) {
    int idx = blockIdx.x * blockDim.x + threadIdx.x;
    if (idx >= K * N) return;
    int k = idx / N, n = idx - k * N;
    float v = 0.f;
    if (mode == 0) {
        if (k < Ksrc) v = src[(size_t)n * Ksrc + k];
    } else {
        int c = k % C, hw = k / C;
        v = src[(size_t)n * Ksrc + c * HW + hw];
    }
    Bm[idx] = (_Float16)v;
}

__global__ void prep_whead_kernel(const float* __restrict__ actor_w,
                                  const float* __restrict__ critic_w,
                                  _Float16* __restrict__ Bm) {
    int idx = blockIdx.x * blockDim.x + threadIdx.x; // K=256, N=32
    if (idx >= 256 * 32) return;
    int k = idx >> 5, n = idx & 31;
    float v = 0.f;
    if (n < 16)       v = actor_w[n * 256 + k];
    else if (n == 16) v = critic_w[k];
    Bm[idx] = (_Float16)v;
}

__global__ void prep_bias_kernel(const float* __restrict__ bih, const float* __restrict__ bhh,
                                 const float* __restrict__ ab, const float* __restrict__ cb,
                                 float* __restrict__ bgate, float* __restrict__ bhead) {
    int idx = blockIdx.x * blockDim.x + threadIdx.x;
    if (idx < 1024) {
        bgate[idx] = bih[idx] + bhh[idx];
    } else if (idx < 1056) {
        int n = idx - 1024;
        float v = 0.f;
        if (n < 16)       v = ab[n];
        else if (n == 16) v = cb[0];
        bhead[n] = v;
    }
}

// B [K][N] -> WMMA fragment order.
__global__ void swizzle_b_kernel(const _Float16* __restrict__ Bm, _Float16* __restrict__ F,
                                 int K, int N) {
    int idx = blockIdx.x * blockDim.x + threadIdx.x;
    if (idx >= K * N) return;
    int numK = K >> 5;
    int f = idx >> 9, r = idx & 511;
    int lane = r >> 4, h = r & 15;
    int nt = f / numK, ks = f - nt * numK;
    int n = nt * 16 + (lane & 15);
    int k = ks * 32 + ((h >> 3) << 4) + ((lane >> 4) << 3) + (h & 7);
    F[idx] = Bm[(size_t)k * N + n];
}

// ---------------------------------------------------------------------------
// conv1: x [N,4,84,84] f32, stride 4 pad 1 -> act1 [N,20,20,32] f16 (NHWC)
// M=400, K=256 (k = c*64 + ky*8 + kx), N=32. Per-wave: one mTile, both nTiles
// (shared A fragment, two independent WMMA chains).
// ---------------------------------------------------------------------------
__global__ __launch_bounds__(256) void conv1_kernel(const float* __restrict__ x,
                                                    const _Float16* __restrict__ w1F,
                                                    const float* __restrict__ bias1,
                                                    _Float16* __restrict__ act1) {
    __shared__ _Float16 sIm[4 * 86 * 86]; // zero-padded image, 57.8 KB
    int img = blockIdx.x, tid = threadIdx.x;
    for (int i = tid; i < 4 * 86 * 86; i += 256) sIm[i] = (_Float16)0.f;
    __syncthreads();
    const float* xi = x + (size_t)img * (4 * 84 * 84);
    for (int i = tid; i < 4 * 84 * 84; i += 256) {
        int c = i / (84 * 84), rem = i - c * 84 * 84;
        int y = rem / 84, xx = rem - y * 84;
        sIm[(c * 86 + y + 1) * 86 + xx + 1] = (_Float16)xi[i];
    }
    __syncthreads();
    int wave = tid >> 5, lane = tid & 31;
    for (int mt = wave; mt < 25; mt += 8) {
        v8f acc0 = {}, acc1 = {};
        int m = mt * 16 + (lane & 15);
        int py = m / 20, px = m - py * 20;
        int khalf = (lane >> 4) * 8;
        for (int ks = 0; ks < 8; ++ks) {
            union { v16h v; unsigned int u[8]; } A;
#pragma unroll
            for (int run = 0; run < 2; ++run) {
                int kG = ks * 32 + run * 16 + khalf;  // multiple of 8 -> kx0 = 0
                int c  = kG >> 6;
                int ky = (kG >> 3) & 7;
                const unsigned int* p =
                    (const unsigned int*)&sIm[(c * 86 + py * 4 + ky) * 86 + px * 4];
#pragma unroll
                for (int i = 0; i < 4; ++i) A.u[run * 4 + i] = p[i];
            }
            v16h fb0 = load_b_frag(w1F + (size_t)ks * 512);
            v16h fb1 = load_b_frag(w1F + (size_t)(8 + ks) * 512);
            acc0 = wmma32f16(A.v, fb0, acc0);
            acc1 = wmma32f16(A.v, fb1, acc1);
        }
        int n = lane & 15;
        float bv0 = bias1[n], bv1 = bias1[16 + n];
#pragma unroll
        for (int r = 0; r < 8; ++r) {
            int mm = mt * 16 + (lane >> 4) * 8 + r;
            size_t base = ((size_t)img * 400 + mm) * 32;
            act1[base + n]      = (_Float16)lrelu(acc0[r] + bv0);
            act1[base + 16 + n] = (_Float16)lrelu(acc1[r] + bv1);
        }
    }
}

// ---------------------------------------------------------------------------
// conv2: act1 [N,20,20,32] -> act2 [N,9,9,64] f16 (NHWC). stride 2, no pad.
// M=81 (6 mTiles, clamped), K=512 (k=(ky*4+kx)*32+c), N=64.
// 12 pair-jobs: (mt, nt-pair) share the A fragment, two WMMA chains.
// ---------------------------------------------------------------------------
__global__ __launch_bounds__(256) void conv2_kernel(const _Float16* __restrict__ act1,
                                                    const _Float16* __restrict__ w2F,
                                                    const float* __restrict__ bias2,
                                                    _Float16* __restrict__ act2) {
    __shared__ _Float16 sA[20 * 20 * 32]; // 25.6 KB
    int img = blockIdx.x, tid = threadIdx.x;
#if USE_TDM
    if (tid == 0) {
        tdm_load_linear(sA, act1 + (size_t)img * 12800, 3200);
        __builtin_amdgcn_s_wait_tensorcnt(0);
    }
#else
    {
        const unsigned int* src = (const unsigned int*)(act1 + (size_t)img * 12800);
        unsigned int* dst = (unsigned int*)sA;
        for (int i = tid; i < 6400; i += 256) dst[i] = src[i];
    }
#endif
    __syncthreads();
    int wave = tid >> 5, lane = tid & 31;
    for (int pj = wave; pj < 12; pj += 8) {   // 6 mTiles x 2 nt-pairs
        int mt = pj >> 1, np = pj & 1;
        int nt0 = np * 2, nt1 = np * 2 + 1;
        v8f acc0 = {}, acc1 = {};
        int m = mt * 16 + (lane & 15); if (m > 80) m = 80;
        int py = m / 9, px = m - py * 9;
        int khalf = (lane >> 4) * 8;
        for (int ks = 0; ks < 16; ++ks) {
            union { v16h v; unsigned int u[8]; } A;
#pragma unroll
            for (int run = 0; run < 2; ++run) {
                int kG  = ks * 32 + run * 16 + khalf;
                int c0  = kG & 31;
                int kyx = kG >> 5;
                int ky = kyx >> 2, kx = kyx & 3;
                const unsigned int* p =
                    (const unsigned int*)&sA[((py * 2 + ky) * 20 + (px * 2 + kx)) * 32 + c0];
#pragma unroll
                for (int i = 0; i < 4; ++i) A.u[run * 4 + i] = p[i];
            }
            v16h fb0 = load_b_frag(w2F + (size_t)(nt0 * 16 + ks) * 512);
            v16h fb1 = load_b_frag(w2F + (size_t)(nt1 * 16 + ks) * 512);
            acc0 = wmma32f16(A.v, fb0, acc0);
            acc1 = wmma32f16(A.v, fb1, acc1);
        }
        int n0 = nt0 * 16 + (lane & 15), n1 = nt1 * 16 + (lane & 15);
        float bv0 = bias2[n0], bv1 = bias2[n1];
#pragma unroll
        for (int r = 0; r < 8; ++r) {
            int mm = mt * 16 + (lane >> 4) * 8 + r;
            if (mm < 81) {
                size_t base = ((size_t)img * 81 + mm) * 64;
                act2[base + n0] = (_Float16)lrelu(acc0[r] + bv0);
                act2[base + n1] = (_Float16)lrelu(acc1[r] + bv1);
            }
        }
    }
}

// ---------------------------------------------------------------------------
// conv3: act2 [N,9,9,64] -> act3 [N,64,7,7] f16 (NCHW flat, matches fc_w K-order)
// M=49 (4 mTiles, clamped), K=576 (k=(ky*3+kx)*64+c), N=64. 8 pair-jobs.
// ---------------------------------------------------------------------------
__global__ __launch_bounds__(256) void conv3_kernel(const _Float16* __restrict__ act2,
                                                    const _Float16* __restrict__ w3F,
                                                    const float* __restrict__ bias3,
                                                    _Float16* __restrict__ act3) {
    __shared__ _Float16 sA[81 * 64]; // 10.4 KB
    int img = blockIdx.x, tid = threadIdx.x;
#if USE_TDM
    if (tid == 0) {
        tdm_load_linear(sA, act2 + (size_t)img * 5184, 1296);
        __builtin_amdgcn_s_wait_tensorcnt(0);
    }
#else
    {
        const unsigned int* src = (const unsigned int*)(act2 + (size_t)img * 5184);
        unsigned int* dst = (unsigned int*)sA;
        for (int i = tid; i < 2592; i += 256) dst[i] = src[i];
    }
#endif
    __syncthreads();
    int wave = tid >> 5, lane = tid & 31;
    {
        int pj = wave;                         // 8 pair-jobs, one per wave
        int mt = pj >> 1, np = pj & 1;
        int nt0 = np * 2, nt1 = np * 2 + 1;
        v8f acc0 = {}, acc1 = {};
        int m = mt * 16 + (lane & 15); if (m > 48) m = 48;
        int py = m / 7, px = m - py * 7;
        int khalf = (lane >> 4) * 8;
        for (int ks = 0; ks < 18; ++ks) {
            union { v16h v; unsigned int u[8]; } A;
#pragma unroll
            for (int run = 0; run < 2; ++run) {
                int kG  = ks * 32 + run * 16 + khalf;
                int c0  = kG & 63;
                int kyx = kG >> 6;
                int ky = kyx / 3, kx = kyx - ky * 3;
                const unsigned int* p =
                    (const unsigned int*)&sA[((py + ky) * 9 + (px + kx)) * 64 + c0];
#pragma unroll
                for (int i = 0; i < 4; ++i) A.u[run * 4 + i] = p[i];
            }
            v16h fb0 = load_b_frag(w3F + (size_t)(nt0 * 18 + ks) * 512);
            v16h fb1 = load_b_frag(w3F + (size_t)(nt1 * 18 + ks) * 512);
            acc0 = wmma32f16(A.v, fb0, acc0);
            acc1 = wmma32f16(A.v, fb1, acc1);
        }
        int n0 = nt0 * 16 + (lane & 15), n1 = nt1 * 16 + (lane & 15);
        float bv0 = bias3[n0], bv1 = bias3[n1];
#pragma unroll
        for (int r = 0; r < 8; ++r) {
            int mm = mt * 16 + (lane >> 4) * 8 + r;
            if (mm < 49) {
                size_t base = (size_t)img * 3136 + mm;
                act3[base + n0 * 49] = (_Float16)lrelu(acc0[r] + bv0);
                act3[base + n1 * 49] = (_Float16)lrelu(acc1[r] + bv1);
            }
        }
    }
}

// ---------------------------------------------------------------------------
// Generic 64x64-tile WMMA GEMM: C = act( A[M][K]_f16 * Bfrag + bias ).
// M multiple of 64. N guarded (heads use N=32). Output f16 (Ch) or f32 (Cf).
// Each wave: one mTile + an nt-pair sharing the A fragment (2 WMMA chains).
// ---------------------------------------------------------------------------
__global__ __launch_bounds__(256) void gemm64_kernel(const _Float16* __restrict__ A, int lda,
                                                     const _Float16* __restrict__ Bfrag, int numK,
                                                     const float* __restrict__ bias, int actMode,
                                                     float* __restrict__ Cf,
                                                     _Float16* __restrict__ Ch,
                                                     int ldc, int N) {
    __shared__ _Float16 sA[64 * 32]; // A chunk per k-step, 4 KB
    int m0 = blockIdx.x * 64, n0 = blockIdx.y * 64;
    int tid = threadIdx.x, wave = tid >> 5, lane = tid & 31;
    int mt = wave >> 1, np = wave & 1;
    int ng0 = (n0 >> 4) + np * 2, ng1 = ng0 + 1;
    bool use0 = (ng0 * 16) < N, use1 = (ng1 * 16) < N;
    v8f acc0 = {}, acc1 = {};
    for (int ks = 0; ks < numK; ++ks) {
        __syncthreads();
        for (int i = tid; i < 1024; i += 256) {
            int row = i >> 4, cw = i & 15;
            ((unsigned int*)sA)[i] =
                *(const unsigned int*)&A[(size_t)(m0 + row) * lda + ks * 32 + cw * 2];
        }
        __syncthreads();
        v16h a = load_a_frag_lds(sA, 32, mt * 16, 0);
        if (use0) {
            v16h b0 = load_b_frag(Bfrag + (size_t)(ng0 * numK + ks) * 512);
            if (ks + 1 < numK)
                __builtin_prefetch(Bfrag + (size_t)(ng0 * numK + ks + 1) * 512, 0, 1);
            acc0 = wmma32f16(a, b0, acc0);
        }
        if (use1) {
            v16h b1 = load_b_frag(Bfrag + (size_t)(ng1 * numK + ks) * 512);
            if (ks + 1 < numK)
                __builtin_prefetch(Bfrag + (size_t)(ng1 * numK + ks + 1) * 512, 0, 1);
            acc1 = wmma32f16(a, b1, acc1);
        }
    }
#pragma unroll
    for (int jj = 0; jj < 2; ++jj) {
        if (jj == 0 && !use0) continue;
        if (jj == 1 && !use1) continue;
        int n = (jj ? ng1 : ng0) * 16 + (lane & 15);
        float bv = bias ? bias[n] : 0.f;
        v8f acc = jj ? acc1 : acc0;
#pragma unroll
        for (int r = 0; r < 8; ++r) {
            int m = m0 + mt * 16 + (lane >> 4) * 8 + r;
            float v = acc[r] + bv;
            if (actMode == 1) v = lrelu(v);
            if (Ch) Ch[(size_t)m * ldc + n] = (_Float16)v;
            else    Cf[(size_t)m * ldc + n] = v;
        }
    }
}

// Fill LSTM-input columns 512..543: reward, one-hot(last_action), zero pad.
__global__ void lstm_fill_kernel(const float* __restrict__ reward,
                                 const int* __restrict__ last_action,
                                 _Float16* __restrict__ xlstm) {
    int idx = blockIdx.x * blockDim.x + threadIdx.x;
    if (idx >= 2560 * 32) return;
    int row = idx >> 5, col = idx & 31;
    _Float16 v = (_Float16)0.f;
    if (col == 0)        v = (_Float16)reward[row];
    else if (col <= 16)  v = (_Float16)((last_action[row] == (col - 1)) ? 1.f : 0.f);
    xlstm[(size_t)row * 544 + 512 + col] = v;
}

// ---------------------------------------------------------------------------
// LSTM recurrence: single workgroup, 512 threads = 16 waves (wave32).
// Dynamic LDS: h f16 [32][256] + c f32 [32][256] + gates f32 [32][1024] =
// 176KB — only feasible on gfx1250's 320KB WGP LDS.
// Per step: 32 quad-jobs (mt, 4 consecutive nTiles) share the A fragment and
// run 4 independent WMMA chains to hide the F16 WMMA hazard.
// ---------------------------------------------------------------------------
__global__ __launch_bounds__(512) void lstm_kernel(const float* __restrict__ gates_x,
                                                   const _Float16* __restrict__ whhF,
                                                   const unsigned char* __restrict__ dones,
                                                   const float* __restrict__ hx,
                                                   _Float16* __restrict__ hsf16) {
    extern __shared__ unsigned char smem[];
    _Float16* sH = (_Float16*)smem;                          // 8192 halves
    float*    sC = (float*)(smem + 16384);                   // 8192 floats
    float*    sG = (float*)(smem + 16384 + 32768);           // 32768 floats
    int tid = threadIdx.x, wave = tid >> 5, lane = tid & 31;
    for (int i = tid; i < 8192; i += 512) {
        sH[i] = (_Float16)hx[i];
        sC[i] = hx[8192 + i];
    }
    __syncthreads();
    for (int t = 0; t < 80; ++t) {
        for (int i = tid; i < 8192; i += 512) {
            int b = i >> 8;
            if (dones[t * 32 + b]) { sH[i] = (_Float16)0.f; sC[i] = 0.f; }
        }
        __syncthreads();
        for (int jp = wave; jp < 32; jp += 16) {  // 2 mTiles x 16 nt-quads
            int mt = jp >> 4, ntq = jp & 15;
            v8f ac[4] = {{}, {}, {}, {}};
#pragma unroll 1
            for (int ks = 0; ks < 8; ++ks) {
                v16h a = load_a_frag_lds(sH, 256, mt * 16, ks * 32);
#pragma unroll
                for (int q = 0; q < 4; ++q) {
                    v16h b = load_b_frag(whhF + (size_t)((ntq * 4 + q) * 8 + ks) * 512);
                    ac[q] = wmma32f16(a, b, ac[q]);
                }
            }
#pragma unroll
            for (int q = 0; q < 4; ++q) {
                int jc = (ntq * 4 + q) * 16 + (lane & 15);
#pragma unroll
                for (int r = 0; r < 8; ++r) {
                    int b = mt * 16 + (lane >> 4) * 8 + r;
                    sG[b * 1024 + jc] = ac[q][r] + gates_x[((size_t)t * 32 + b) * 1024 + jc];
                }
            }
        }
        __syncthreads();
        for (int i = tid; i < 8192; i += 512) {
            int b = i >> 8, u = i & 255;
            float gi = sG[b * 1024 + u];
            float gf = sG[b * 1024 + 256 + u];
            float gg = sG[b * 1024 + 512 + u];
            float go = sG[b * 1024 + 768 + u];
            float c  = sigm(gf) * sC[i] + sigm(gi) * tanhf(gg);
            float hh = sigm(go) * tanhf(c);
            sC[i] = c;
            sH[i] = (_Float16)hh;
            hsf16[((size_t)t * 32 + b) * 256 + u] = (_Float16)hh;
        }
        __syncthreads();
    }
}

// Scatter head GEMM result into d_out: logits (NaN-fixed) then values.
__global__ void finalize_kernel(const float* __restrict__ head_out, float* __restrict__ out) {
    int idx = blockIdx.x * blockDim.x + threadIdx.x;
    if (idx >= 2560 * 17) return;
    if (idx < 40960) {
        int row = idx >> 4, a = idx & 15;
        float v = head_out[row * 32 + a];
        if (v != v) v = 1e-12f;
        out[idx] = v;
    } else {
        int row = idx - 40960;
        out[40960 + row] = head_out[row * 32 + 16];
    }
}

// ---------------------------------------------------------------------------
extern "C" void kernel_launch(void* const* d_in, const int* in_sizes, int n_in,
                              void* d_out, int out_size, void* d_ws, size_t ws_size,
                              hipStream_t stream) {
    (void)in_sizes; (void)n_in; (void)out_size;
    const float* x           = (const float*)d_in[0];
    const int*   last_action = (const int*)d_in[1];
    const float* reward      = (const float*)d_in[2];
    const unsigned char* dones = (const unsigned char*)d_in[3];
    const float* hx   = (const float*)d_in[4];
    const float* c1w  = (const float*)d_in[5];  const float* c1b = (const float*)d_in[6];
    const float* c2w  = (const float*)d_in[7];  const float* c2b = (const float*)d_in[8];
    const float* c3w  = (const float*)d_in[9];  const float* c3b = (const float*)d_in[10];
    const float* fcw  = (const float*)d_in[11]; const float* fcb = (const float*)d_in[12];
    const float* wih  = (const float*)d_in[13]; const float* whh = (const float*)d_in[14];
    const float* bih  = (const float*)d_in[15]; const float* bhh = (const float*)d_in[16];
    const float* aw   = (const float*)d_in[17]; const float* ab  = (const float*)d_in[18];
    const float* cw   = (const float*)d_in[19]; const float* cb  = (const float*)d_in[20];
    float* out = (float*)d_out;

    unsigned char* ws = (unsigned char*)d_ws;
    size_t off = 0;
    auto take = [&](size_t bytes) -> unsigned char* {
        unsigned char* p = ws + off;
        off = (off + bytes + 255) & ~(size_t)255;
        return p;
    };
    // activations (total ws use ~133 MB)
    _Float16* act1   = (_Float16*)take(65536000);  // [2560,400,32]
    _Float16* act2   = (_Float16*)take(26542080);  // [2560,81,64]
    _Float16* act3   = (_Float16*)take(16056320);  // [2560,3136] NCHW
    _Float16* xlstm  = (_Float16*)take(2785280);   // [2560,544]
    float*    gatesx = (float*)take(10485760);     // [2560,1024]
    _Float16* hsf16  = (_Float16*)take(1310720);   // [2560,256]
    float*    headout= (float*)take(327680);       // [2560,32]
    // weights (Bmat + swizzled fragments)
    _Float16* w1B = (_Float16*)take(16384);    _Float16* w1F = (_Float16*)take(16384);
    _Float16* w2B = (_Float16*)take(65536);    _Float16* w2F = (_Float16*)take(65536);
    _Float16* w3B = (_Float16*)take(73728);    _Float16* w3F = (_Float16*)take(73728);
    _Float16* wfB = (_Float16*)take(3211264);  _Float16* wfF = (_Float16*)take(3211264);
    _Float16* wiB = (_Float16*)take(1114112);  _Float16* wiF = (_Float16*)take(1114112);
    _Float16* whB = (_Float16*)take(524288);   _Float16* whF = (_Float16*)take(524288);
    _Float16* whdB= (_Float16*)take(16384);    _Float16* whdF= (_Float16*)take(16384);
    float* bgate = (float*)take(4096);
    float* bhead = (float*)take(128);
    (void)ws_size;

    auto blks = [](int total) { return (total + 255) / 256; };

    // --- weight prep (runs every call; tiny) ---
    prep_b_kernel<<<blks(256*32),   256, 0, stream>>>(c1w, w1B, 256,  32,  256,  0,  0, 0);
    prep_b_kernel<<<blks(512*64),   256, 0, stream>>>(c2w, w2B, 512,  64,  512,  32, 16, 1);
    prep_b_kernel<<<blks(576*64),   256, 0, stream>>>(c3w, w3B, 576,  64,  576,  64, 9,  1);
    prep_b_kernel<<<blks(3136*512), 256, 0, stream>>>(fcw, wfB, 3136, 512, 3136, 0,  0, 0);
    prep_b_kernel<<<blks(544*1024), 256, 0, stream>>>(wih, wiB, 544, 1024, 529,  0,  0, 0);
    prep_b_kernel<<<blks(256*1024), 256, 0, stream>>>(whh, whB, 256, 1024, 256,  0,  0, 0);
    prep_whead_kernel<<<blks(256*32), 256, 0, stream>>>(aw, cw, whdB);
    prep_bias_kernel<<<blks(1056), 256, 0, stream>>>(bih, bhh, ab, cb, bgate, bhead);

    swizzle_b_kernel<<<blks(256*32),   256, 0, stream>>>(w1B, w1F, 256,  32);
    swizzle_b_kernel<<<blks(512*64),   256, 0, stream>>>(w2B, w2F, 512,  64);
    swizzle_b_kernel<<<blks(576*64),   256, 0, stream>>>(w3B, w3F, 576,  64);
    swizzle_b_kernel<<<blks(3136*512), 256, 0, stream>>>(wfB, wfF, 3136, 512);
    swizzle_b_kernel<<<blks(544*1024), 256, 0, stream>>>(wiB, wiF, 544, 1024);
    swizzle_b_kernel<<<blks(256*1024), 256, 0, stream>>>(whB, whF, 256, 1024);
    swizzle_b_kernel<<<blks(256*32),   256, 0, stream>>>(whdB, whdF, 256, 32);

    // --- conv tower (one image per workgroup, WMMA implicit GEMM) ---
    conv1_kernel<<<2560, 256, 0, stream>>>(x, w1F, c1b, act1);
    conv2_kernel<<<2560, 256, 0, stream>>>(act1, w2F, c2b, act2);
    conv3_kernel<<<2560, 256, 0, stream>>>(act2, w3F, c3b, act3);

    // --- FC: [2560,3136] @ [3136,512] -> lrelu -> xlstm cols 0..511 ---
    gemm64_kernel<<<dim3(40, 8), 256, 0, stream>>>(act3, 3136, wfF, 98,
                                                   fcb, 1, nullptr, xlstm, 544, 512);
    lstm_fill_kernel<<<blks(2560*32), 256, 0, stream>>>(reward, last_action, xlstm);

    // --- LSTM input projection for all T: [2560,544] @ [544,1024] + bgate ---
    gemm64_kernel<<<dim3(40, 16), 256, 0, stream>>>(xlstm, 544, wiF, 17,
                                                    bgate, 0, gatesx, nullptr, 1024, 1024);

    // --- sequential LSTM recurrence (176KB dynamic LDS) ---
    lstm_kernel<<<1, 512, 180224, stream>>>(gatesx, whF, dones, hx, hsf16);

    // --- heads: [2560,256] @ [256,32] (actor cols 0..15, critic col 16) ---
    gemm64_kernel<<<dim3(40, 1), 256, 0, stream>>>(hsf16, 256, whdF, 8,
                                                   bhead, 0, headout, nullptr, 32, 32);
    finalize_kernel<<<blks(2560*17), 256, 0, stream>>>(headout, out);
}
